// GatedAttention_85581518340358
// MI455X (gfx1250) — matrix operand order repaired and checked by reading, not compiled, not run on hardware
//
#include <hip/hip_runtime.h>
#include <hip/hip_bf16.h>

typedef __attribute__((ext_vector_type(16))) __bf16 v16bf;
typedef __attribute__((ext_vector_type(8)))  __bf16 v8bf;
typedef __attribute__((ext_vector_type(8)))  float  v8f;

#define B_     2
#define NQ_    1024
#define NKV_   2048   // NQ + NC
#define QD_    1024   // QUERY_DIM == CONTEXT_DIM
#define INNER_ 512    // HEADS * DIM_HEAD
#define HEADS_ 8
#define DH_    64

static __device__ __forceinline__ v8f zerov8() {
  v8f z = {0.f,0.f,0.f,0.f,0.f,0.f,0.f,0.f};
  return z;
}

// A-fragment (16x32 bf16, M=row striped in lane): lanes 0-15 hold k=0..7 & 16..23,
// lanes 16-31 hold k=8..15 & 24..31, of row (lane%16).
static __device__ __forceinline__ v16bf load_fragA(const __bf16* row, int kbase, int hi) {
  const int off = hi ? 8 : 0;
  v8bf a = *(const v8bf*)(row + kbase + off);
  v8bf b = *(const v8bf*)(row + kbase + off + 16);
  return __builtin_shufflevector(a, b, 0,1,2,3,4,5,6,7,8,9,10,11,12,13,14,15);
}

// B-fragment (32x16 bf16, N=col striped across lanes): lanes 0-15 hold k=0..7 & 16..23,
// lanes 16-31 hold k=8..15 & 24..31, of column (lane%16). `col` points at N-major
// (K-contiguous) storage of that column.
static __device__ __forceinline__ v16bf load_fragB(const __bf16* col, int kbase, int hi) {
  const int off = hi ? 8 : 0;
  v8bf a = *(const v8bf*)(col + kbase + off);
  v8bf b = *(const v8bf*)(col + kbase + off + 16);
  return __builtin_shufflevector(a, b, 0,1,2,3,4,5,6,7,8,9,10,11,12,13,14,15);
}

static __device__ __forceinline__ v8f wmma_bf16(v16bf a, v16bf b, v8f c) {
  return __builtin_amdgcn_wmma_f32_16x16x32_bf16(false, a, false, b, (short)0, c, false, false);
}

// ---------------- kernel 1: concat(x, context) -> bf16 ----------------
__global__ void ga_concat_bf16(const float* __restrict__ x,
                               const float* __restrict__ ctx,
                               __bf16* __restrict__ in_cat) {
  size_t idx = (size_t)blockIdx.x * blockDim.x + threadIdx.x;
  const size_t total = (size_t)B_ * NKV_ * QD_;
  if (idx >= total) return;
  size_t b = idx / ((size_t)NKV_ * QD_);
  size_t rem = idx - b * (size_t)NKV_ * QD_;
  size_t r = rem / QD_, c = rem % QD_;
  float v = (r < NQ_) ? x[(b * NQ_ + r) * QD_ + c]
                      : ctx[(b * NQ_ + (r - NQ_)) * QD_ + c];
  in_cat[idx] = (__bf16)v;
}

// ---------------- kernel 2: W (KxN f32) -> WT (NxK bf16) ----------------
__global__ void ga_transpose_bf16(const float* __restrict__ W,
                                  __bf16* __restrict__ WT, int K, int N) {
  size_t idx = (size_t)blockIdx.x * blockDim.x + threadIdx.x;
  if (idx >= (size_t)K * N) return;
  int k = (int)(idx % K);
  int n = (int)(idx / K);
  WT[(size_t)n * K + k] = (__bf16)W[(size_t)k * N + n];
}

// -------- kernel 3/4: dual-weight projection  out = in_cat @ [W0|W1] --------
// 32x64 tile per wave (2 row-subtiles share every B fragment).
// Rows < NQ_ use W0T (x-side), rows >= NQ_ use W1T (context-side).
// transposed==0 : out row-major (b, NKV_, INNER_)   -> qc
// transposed==1 : out inner-major (b, INNER_, NKV_) -> vT
__global__ void ga_proj_dual(const __bf16* __restrict__ in_cat,
                             const __bf16* __restrict__ W0T,
                             const __bf16* __restrict__ W1T,
                             __bf16* __restrict__ out, int transposed) {
  const int lane = threadIdx.x & 31, hi = lane >> 4, l16 = lane & 15;
  const int w = threadIdx.x >> 5;
  const int b = blockIdx.z;
  const int mb = blockIdx.x * 256 + w * 32;   // 32 rows per wave
  const int nb = blockIdx.y * 64;

  const __bf16* Arow0 = in_cat + ((size_t)b * NKV_ + mb + l16) * QD_;
  const __bf16* Arow1 = Arow0 + (size_t)16 * QD_;
  const __bf16* WT = (mb < NQ_) ? W0T : W1T;   // 1024 % 32 == 0: tile never straddles
  const __bf16* Bcol[4];
  #pragma unroll
  for (int t = 0; t < 4; ++t) Bcol[t] = WT + (size_t)(nb + t * 16 + l16) * QD_;

  v8f acc[2][4];
  #pragma unroll
  for (int g = 0; g < 2; ++g)
    #pragma unroll
    for (int t = 0; t < 4; ++t) acc[g][t] = zerov8();

  #pragma unroll 4
  for (int k = 0; k < QD_; k += 32) {
    v16bf a0 = load_fragA(Arow0, k, hi);
    v16bf a1 = load_fragA(Arow1, k, hi);
    #pragma unroll
    for (int t = 0; t < 4; ++t) {
      v16bf bt = load_fragB(Bcol[t], k, hi);
      acc[0][t] = wmma_bf16(a0, bt, acc[0][t]);
      acc[1][t] = wmma_bf16(a1, bt, acc[1][t]);
    }
  }

  if (transposed) {
    #pragma unroll
    for (int g = 0; g < 2; ++g) {
      const int mstart = mb + g * 16 + hi * 8;
      #pragma unroll
      for (int t = 0; t < 4; ++t) {
        const int n = nb + t * 16 + l16;
        v8bf pk;
        #pragma unroll
        for (int r = 0; r < 8; ++r) pk[r] = (__bf16)acc[g][t][r];
        *(v8bf*)(out + ((size_t)b * INNER_ + n) * NKV_ + mstart) = pk;
      }
    }
  } else {
    #pragma unroll
    for (int g = 0; g < 2; ++g)
      #pragma unroll
      for (int t = 0; t < 4; ++t) {
        const int n = nb + t * 16 + l16;
        #pragma unroll
        for (int r = 0; r < 8; ++r) {
          const int m = mb + g * 16 + r + hi * 8;
          out[((size_t)b * NKV_ + m) * INNER_ + n] = (__bf16)acc[g][t][r];
        }
      }
  }
}

// ---------------- kernel 5: flash attention (queries 0..NQ_-1 only) ----------------
// One wave handles 32 queries (2 groups of 16) x all 2048 keys for one (b,h).
// Computes S^T = K @ Q^T: the C-layout of S^T maps directly onto the P A-fragment.
// K A-fragments and V B-fragments are shared across both query groups.
__global__ void ga_attention(const __bf16* __restrict__ qc,
                             const __bf16* __restrict__ vT,
                             const unsigned char* __restrict__ mask,
                             __bf16* __restrict__ attn_out) {
  const int lane = threadIdx.x & 31, hi = lane >> 4, l16 = lane & 15;
  const int w = threadIdx.x >> 5;
  const int qb = blockIdx.x * 256 + w * 32;    // 32 queries per wave
  const int h = blockIdx.y, b = blockIdx.z;
  const float scale = 0.125f;  // DH_^-0.5
  const float NEG = -3.0e38f;

  // Loop-invariant Q B-fragments (Q^T columns = query rows of qc), per group
  const __bf16* Qrow0 = qc + ((size_t)b * NKV_ + qb + l16) * INNER_ + h * DH_;
  const __bf16* Qrow1 = Qrow0 + (size_t)16 * INNER_;
  v16bf bq[2][2];
  bq[0][0] = load_fragB(Qrow0, 0, hi);  bq[0][1] = load_fragB(Qrow0, 32, hi);
  bq[1][0] = load_fragB(Qrow1, 0, hi);  bq[1][1] = load_fragB(Qrow1, 32, hi);

  v8f o[2][4];
  #pragma unroll
  for (int g = 0; g < 2; ++g)
    #pragma unroll
    for (int t = 0; t < 4; ++t) o[g][t] = zerov8();
  float m_run[2] = {NEG, NEG};
  float l_run[2] = {0.f, 0.f};
  const unsigned char* mrow = mask + (size_t)b * NKV_;

  for (int c = 0; c < NKV_; c += 32) {
    const __bf16* Krow0 = qc + ((size_t)b * NKV_ + c + l16) * INNER_ + h * DH_;
    const __bf16* Krow1 = Krow0 + (size_t)16 * INNER_;
    // K fragments, shared by both query groups
    v16bf a00 = load_fragA(Krow0, 0, hi), a01 = load_fragA(Krow0, 32, hi);
    v16bf a10 = load_fragA(Krow1, 0, hi), a11 = load_fragA(Krow1, 32, hi);

    // per-lane key rows: tile0 -> c + r + hi*8, tile1 -> +16 (shared by groups)
    unsigned long long mm0 = *(const unsigned long long*)(mrow + c + hi * 8);
    unsigned long long mm1 = *(const unsigned long long*)(mrow + c + 16 + hi * 8);

    v16bf pa[2];
    float asc[2][8];
    #pragma unroll
    for (int g = 0; g < 2; ++g) {
      // S^T tiles: tile0 = keys c..c+15, tile1 = keys c+16..c+31 (M), queries (N)
      v8f s0 = zerov8(), s1 = zerov8();
      s0 = wmma_bf16(a00, bq[g][0], s0);
      s0 = wmma_bf16(a01, bq[g][1], s0);
      s1 = wmma_bf16(a10, bq[g][0], s1);
      s1 = wmma_bf16(a11, bq[g][1], s1);

      float p0[8], p1[8];
      float cmax = NEG;
      #pragma unroll
      for (int r = 0; r < 8; ++r) {
        float v0 = s0[r] * scale;
        float v1 = s1[r] * scale;
        if (!((mm0 >> (8 * r)) & 0xffull)) v0 = NEG;
        if (!((mm1 >> (8 * r)) & 0xffull)) v1 = NEG;
        p0[r] = v0; p1[r] = v1;
        cmax = fmaxf(cmax, fmaxf(v0, v1));
      }
      // sync running max across the lane pair holding the same query
      cmax = fmaxf(cmax, __shfl_xor(cmax, 16));
      float m_new = fmaxf(m_run[g], cmax);
      float alpha = __expf(m_run[g] - m_new);

      float psum = 0.f;
      #pragma unroll
      for (int r = 0; r < 8; ++r) {
        float e0 = __expf(p0[r] - m_new);
        float e1 = __expf(p1[r] - m_new);
        psum += e0 + e1;
        pa[g][r]     = (__bf16)e0;   // A-frag of P: [tile0 rows, tile1 rows]
        pa[g][r + 8] = (__bf16)e1;
      }
      l_run[g] = l_run[g] * alpha + psum;
      m_run[g] = m_new;

      // broadcast per-query alpha to the O C-layout rows (query r + hi*8)
      #pragma unroll
      for (int r = 0; r < 8; ++r) asc[g][r] = __shfl(alpha, r + hi * 8);
    }

    #pragma unroll
    for (int t = 0; t < 4; ++t) {
      const __bf16* Vrow = vT + ((size_t)b * INNER_ + h * DH_ + t * 16 + l16) * NKV_;
      v16bf bv = load_fragB(Vrow, c, hi);   // shared by both groups
      #pragma unroll
      for (int g = 0; g < 2; ++g) {
        #pragma unroll
        for (int r = 0; r < 8; ++r) o[g][t][r] *= asc[g][r];
        o[g][t] = wmma_bf16(pa[g], bv, o[g][t]);
      }
    }
  }

  #pragma unroll
  for (int g = 0; g < 2; ++g) {
    float l_tot = l_run[g] + __shfl_xor(l_run[g], 16);
    float inv = 1.f / l_tot;
    float invr[8];
    #pragma unroll
    for (int r = 0; r < 8; ++r) invr[r] = __shfl(inv, r + hi * 8);
    #pragma unroll
    for (int t = 0; t < 4; ++t) {
      const int col = h * DH_ + t * 16 + l16;
      #pragma unroll
      for (int r = 0; r < 8; ++r) {
        const int row = qb + g * 16 + r + hi * 8;
        attn_out[((size_t)b * NQ_ + row) * INNER_ + col] = (__bf16)(o[g][t][r] * invr[r]);
      }
    }
  }
}

// ---------------- kernel 6: out = attn @ Wo + bo (f32 result) ----------------
// 32x64 tile per wave.
__global__ void ga_out_proj(const __bf16* __restrict__ attn,
                            const __bf16* __restrict__ WoT,
                            const float* __restrict__ bo,
                            float* __restrict__ out) {
  const int lane = threadIdx.x & 31, hi = lane >> 4, l16 = lane & 15;
  const int w = threadIdx.x >> 5;
  const int b = blockIdx.z;
  const int mb = blockIdx.x * 256 + w * 32;
  const int nb = blockIdx.y * 64;

  const __bf16* Arow0 = attn + ((size_t)b * NQ_ + mb + l16) * INNER_;
  const __bf16* Arow1 = Arow0 + (size_t)16 * INNER_;
  const __bf16* Bcol[4];
  #pragma unroll
  for (int t = 0; t < 4; ++t) Bcol[t] = WoT + (size_t)(nb + t * 16 + l16) * INNER_;

  v8f acc[2][4];
  #pragma unroll
  for (int g = 0; g < 2; ++g)
    #pragma unroll
    for (int t = 0; t < 4; ++t) acc[g][t] = zerov8();

  #pragma unroll 4
  for (int k = 0; k < INNER_; k += 32) {
    v16bf a0 = load_fragA(Arow0, k, hi);
    v16bf a1 = load_fragA(Arow1, k, hi);
    #pragma unroll
    for (int t = 0; t < 4; ++t) {
      v16bf bt = load_fragB(Bcol[t], k, hi);
      acc[0][t] = wmma_bf16(a0, bt, acc[0][t]);
      acc[1][t] = wmma_bf16(a1, bt, acc[1][t]);
    }
  }

  #pragma unroll
  for (int g = 0; g < 2; ++g)
    #pragma unroll
    for (int t = 0; t < 4; ++t) {
      const int n = nb + t * 16 + l16;
      const float bias = bo[n];
      #pragma unroll
      for (int r = 0; r < 8; ++r) {
        const int m = mb + g * 16 + r + hi * 8;
        out[((size_t)b * NQ_ + m) * QD_ + n] = acc[g][t][r] + bias;
      }
    }
}

extern "C" void kernel_launch(void* const* d_in, const int* in_sizes, int n_in,
                              void* d_out, int out_size, void* d_ws, size_t ws_size,
                              hipStream_t stream) {
  (void)in_sizes; (void)n_in; (void)out_size;
  const float* x    = (const float*)d_in[0];
  const float* ctx  = (const float*)d_in[1];
  const unsigned char* mask = (const unsigned char*)d_in[2];
  const float* Wq   = (const float*)d_in[3];
  const float* Wk   = (const float*)d_in[4];
  const float* Wv   = (const float*)d_in[5];
  const float* Wvs  = (const float*)d_in[6];
  const float* Wo   = (const float*)d_in[7];
  const float* bo   = (const float*)d_in[8];
  float* out = (float*)d_out;

  // workspace carve-up (bf16 buffers)
  char* p = (char*)d_ws;
  __bf16* in_cat = (__bf16*)p; p += (size_t)B_ * NKV_ * QD_ * 2;      // 8 MB
  __bf16* WqT  = (__bf16*)p;   p += (size_t)INNER_ * QD_ * 2;          // 1 MB
  __bf16* WkT  = (__bf16*)p;   p += (size_t)INNER_ * QD_ * 2;
  __bf16* WvsT = (__bf16*)p;   p += (size_t)INNER_ * QD_ * 2;
  __bf16* WvT  = (__bf16*)p;   p += (size_t)INNER_ * QD_ * 2;
  __bf16* WoT  = (__bf16*)p;   p += (size_t)QD_ * INNER_ * 2;
  __bf16* qc   = (__bf16*)p;   p += (size_t)B_ * NKV_ * INNER_ * 2;    // 4 MB
  __bf16* vT   = (__bf16*)p;   p += (size_t)B_ * INNER_ * NKV_ * 2;    // 4 MB
  __bf16* attn = (__bf16*)p;   p += (size_t)B_ * NQ_ * INNER_ * 2;     // 2 MB
  if ((size_t)(p - (char*)d_ws) > ws_size) return;  // workspace too small

  // 1) concat + bf16 convert
  {
    size_t total = (size_t)B_ * NKV_ * QD_;
    ga_concat_bf16<<<(unsigned)((total + 255) / 256), 256, 0, stream>>>(x, ctx, in_cat);
  }
  // 2) weight transposes
  ga_transpose_bf16<<<(QD_ * INNER_) / 256, 256, 0, stream>>>(Wq,  WqT,  QD_,    INNER_);
  ga_transpose_bf16<<<(QD_ * INNER_) / 256, 256, 0, stream>>>(Wk,  WkT,  QD_,    INNER_);
  ga_transpose_bf16<<<(QD_ * INNER_) / 256, 256, 0, stream>>>(Wvs, WvsT, QD_,    INNER_);
  ga_transpose_bf16<<<(QD_ * INNER_) / 256, 256, 0, stream>>>(Wv,  WvT,  QD_,    INNER_);
  ga_transpose_bf16<<<(INNER_ * QD_) / 256, 256, 0, stream>>>(Wo,  WoT,  INNER_, QD_);

  // 3) qc = [x@Wq ; ctx@Wk]  (row-major, also serves as K)
  ga_proj_dual<<<dim3(NKV_ / 256, INNER_ / 64, B_), 256, 0, stream>>>(in_cat, WqT, WkT, qc, 0);
  // 4) v = [x@Wv_self ; ctx@Wv]  stored transposed (inner-major)
  ga_proj_dual<<<dim3(NKV_ / 256, INNER_ / 64, B_), 256, 0, stream>>>(in_cat, WvsT, WvT, vT, 1);
  // 5) attention over query rows only (32 queries per wave)
  ga_attention<<<dim3(NQ_ / 256, HEADS_, B_), 256, 0, stream>>>(qc, vT, mask, attn);
  // 6) output projection + bias
  ga_out_proj<<<dim3(NQ_ / 256, QD_ / 64, B_), 256, 0, stream>>>(attn, WoT, bo, out);
}
